// BiaffinePairing_24129126269150
// MI455X (gfx1250) — matrix operand order repaired and checked by reading, not compiled, not run on hardware
//
#include <hip/hip_runtime.h>
#include <hip/hip_bf16.h>

#define H 1024
#define NROWS 4096

typedef __attribute__((ext_vector_type(16))) __bf16 v16bf;
typedef __attribute__((ext_vector_type(8)))  __bf16 v8bf;
typedef __attribute__((ext_vector_type(8)))  float  v8f;

// ---------- bf16 split helpers (round-to-nearest-even) ----------
__device__ __forceinline__ unsigned short f32_to_bf16_rne(float f) {
  unsigned u = __float_as_uint(f);
  u += 0x7FFFu + ((u >> 16) & 1u);
  return (unsigned short)(u >> 16);
}
__device__ __forceinline__ float bf16_bits_to_f32(unsigned short h) {
  return __uint_as_float(((unsigned)h) << 16);
}

// ---------- W transpose + split: Wt_hi/lo[n][k] = split(W[k][n]) ----------
__global__ __launch_bounds__(256) void convert_w_transpose(
    const float* __restrict__ W,
    unsigned short* __restrict__ Whi, unsigned short* __restrict__ Wlo) {
  __shared__ float tile[32][33];
  const int k0 = blockIdx.x * 32;
  const int n0 = blockIdx.y * 32;
  const int tx = threadIdx.x;   // 0..31
  const int ty = threadIdx.y;   // 0..7
#pragma unroll
  for (int i = 0; i < 32; i += 8)
    tile[ty + i][tx] = W[(size_t)(k0 + ty + i) * H + (n0 + tx)];
  __syncthreads();
#pragma unroll
  for (int i = 0; i < 32; i += 8) {
    float x = tile[tx][ty + i];              // = W[k0+tx][n0+ty+i]
    unsigned short h = f32_to_bf16_rne(x);
    unsigned short l = f32_to_bf16_rne(x - bf16_bits_to_f32(h));
    size_t o = (size_t)(n0 + ty + i) * H + (k0 + tx);
    Whi[o] = h;
    Wlo[o] = l;
  }
}

// ---------- row split + fused rank-1 dot: vec[row] = X[row,:] . u ----------
__global__ __launch_bounds__(256) void convert_rows_dot(
    const float* __restrict__ X, const float* __restrict__ u,
    unsigned short* __restrict__ Xhi, unsigned short* __restrict__ Xlo,
    float* __restrict__ vec) {
  __shared__ float red[256];
  const int row = blockIdx.x;
  const float* xr = X + (size_t)row * H;
  float part = 0.0f;
#pragma unroll
  for (int c = threadIdx.x; c < H; c += 256) {
    float x = xr[c];
    unsigned short h = f32_to_bf16_rne(x);
    unsigned short l = f32_to_bf16_rne(x - bf16_bits_to_f32(h));
    size_t o = (size_t)row * H + c;
    Xhi[o] = h;
    Xlo[o] = l;
    part += x * u[c];
  }
  red[threadIdx.x] = part;
  __syncthreads();
#pragma unroll
  for (int s = 128; s > 0; s >>= 1) {
    if ((int)threadIdx.x < s) red[threadIdx.x] += red[threadIdx.x + s];
    __syncthreads();
  }
  if (threadIdx.x == 0) vec[row] = red[0];
}

// ---------- WMMA fragment loaders (contiguous k order, same for A and B) ----------
// A-matrix 16x32 bf16: lane = half*16 + m; element s = K[half*16 + s]
// -> one contiguous 32B chunk (matches B fragment k order).
__device__ __forceinline__ v16bf load_a_frag(const __bf16* p) {
  v8bf c0 = *(const v8bf*)(p);
  v8bf c1 = *(const v8bf*)(p + 8);
  return __builtin_shufflevector(c0, c1, 0, 1, 2, 3, 4, 5, 6, 7,
                                 8, 9, 10, 11, 12, 13, 14, 15);
}
// B-matrix 32x16 bf16: lane = half*16 + n; VGPR v = K{half*16+2v, +2v+1}
// -> one contiguous 32B chunk per lane when B-rows are contiguous in memory.
__device__ __forceinline__ v16bf load_b_frag(const __bf16* p) {
  return *(const v16bf*)(p);
}

// ---------- bf16x3 emulated-f32 GEMM:  C = Arow * Brow^T ----------
// A: M x K row-major hi/lo.  Bsrc: N x K row-major hi/lo (B[k][n]=Bsrc[n][k]).
// Wave tile 32x64 (2 M-tiles x 4 N-tiles), 8 waves as 2(M) x 4(N) -> 64x256 block.
// EPI==0: split f32 acc -> Chi/Clo (bf16 hi/lo), for the intermediate TW.
// EPI==1: Cf[i][j] = acc + rowv[i] + colv[j] + *bias   (fused biaffine epilogue)
template <int EPI>
__global__ __launch_bounds__(256) void gemm_bf16x3(
    const __bf16* __restrict__ Ahi, const __bf16* __restrict__ Alo, int lda,
    const __bf16* __restrict__ Bhi, const __bf16* __restrict__ Blo, int ldb,
    int K,
    float* __restrict__ Cf, int ldc,
    unsigned short* __restrict__ Chi, unsigned short* __restrict__ Clo,
    const float* __restrict__ rowv, const float* __restrict__ colv,
    const float* __restrict__ bias) {
  const int lane   = threadIdx.x & 31;
  const int wave   = threadIdx.x >> 5;
  const int wm     = wave & 1;   // 2 waves along M
  const int wn     = wave >> 1;  // 4 waves along N
  const int lane15 = lane & 15;
  const int half   = lane >> 4;

  const int m0 = blockIdx.x * 64 + wm * 32;    // wave's 32x64 output origin
  const int n0 = blockIdx.y * 256 + wn * 64;

  v8f acc[2][4] = {};

  // Per-lane base pointers (K advances by +k).
  const __bf16* aB[2][2];
  const __bf16* bB[4][2];
#pragma unroll
  for (int mt = 0; mt < 2; ++mt) {
    size_t r = (size_t)(m0 + mt * 16 + lane15) * lda + half * 16;
    aB[mt][0] = Ahi + r;
    aB[mt][1] = Alo + r;
  }
#pragma unroll
  for (int nt = 0; nt < 4; ++nt) {
    size_t r = (size_t)(n0 + nt * 16 + lane15) * ldb + half * 16;
    bB[nt][0] = Bhi + r;
    bB[nt][1] = Blo + r;
  }

  for (int k = 0; k < K; k += 32) {
    if (k + 32 < K) {  // uniform branch: prefetch next K-block
      __builtin_prefetch(aB[0][0] + k + 32, 0, 1);
      __builtin_prefetch(aB[1][0] + k + 32, 0, 1);
      __builtin_prefetch(bB[0][0] + k + 32, 0, 1);
      __builtin_prefetch(bB[2][0] + k + 32, 0, 1);
    }
    v16bf ah[2], al[2], bh[4], bl[4];
#pragma unroll
    for (int mt = 0; mt < 2; ++mt) {
      ah[mt] = load_a_frag(aB[mt][0] + k);
      al[mt] = load_a_frag(aB[mt][1] + k);
    }
#pragma unroll
    for (int nt = 0; nt < 4; ++nt) {
      bh[nt] = load_b_frag(bB[nt][0] + k);
      bl[nt] = load_b_frag(bB[nt][1] + k);
    }
    // mt outer: A fragments stay live across 12 consecutive WMMAs (operand reuse)
#pragma unroll
    for (int mt = 0; mt < 2; ++mt)
#pragma unroll
      for (int nt = 0; nt < 4; ++nt) {
        // bf16x3: hi*hi + hi*lo + lo*hi  (lo*lo dropped, below f32 ulp)
        acc[mt][nt] = __builtin_amdgcn_wmma_f32_16x16x32_bf16(
            false, ah[mt], false, bh[nt], (short)0, acc[mt][nt], false, false);
        acc[mt][nt] = __builtin_amdgcn_wmma_f32_16x16x32_bf16(
            false, ah[mt], false, bl[nt], (short)0, acc[mt][nt], false, false);
        acc[mt][nt] = __builtin_amdgcn_wmma_f32_16x16x32_bf16(
            false, al[mt], false, bh[nt], (short)0, acc[mt][nt], false, false);
      }
  }

  // Epilogue.  C/D layout: VGPR r -> row (r + 8*half), col = lane15.
  float bv = 0.0f;
  if constexpr (EPI == 1) bv = *bias;
#pragma unroll
  for (int mt = 0; mt < 2; ++mt) {
    const int rowbase = m0 + mt * 16 + half * 8;
#pragma unroll
    for (int nt = 0; nt < 4; ++nt) {
      const int col = n0 + nt * 16 + lane15;
      v8f c = acc[mt][nt];
      if constexpr (EPI == 1) {
        const float cadd = colv[col] + bv;
#pragma unroll
        for (int r = 0; r < 8; ++r) {
          const int row = rowbase + r;
          Cf[(size_t)row * ldc + col] = c[r] + rowv[row] + cadd;
        }
      } else {
#pragma unroll
        for (int r = 0; r < 8; ++r) {
          const int row = rowbase + r;
          float x = c[r];
          unsigned short h = f32_to_bf16_rne(x);
          unsigned short l = f32_to_bf16_rne(x - bf16_bits_to_f32(h));
          Chi[(size_t)row * ldc + col] = h;
          Clo[(size_t)row * ldc + col] = l;
        }
      }
    }
  }
}

extern "C" void kernel_launch(void* const* d_in, const int* in_sizes, int n_in,
                              void* d_out, int out_size, void* d_ws, size_t ws_size,
                              hipStream_t stream) {
  (void)in_sizes; (void)n_in; (void)out_size; (void)ws_size;
  const float* T = (const float*)d_in[0];   // 4096 x 1024
  const float* A = (const float*)d_in[1];   // 4096 x 1024
  const float* W = (const float*)d_in[2];   // 1024 x 1024
  const float* U = (const float*)d_in[3];   // 2048 (u_t | u_a)
  const float* b = (const float*)d_in[4];   // 1

  // Workspace carve (256B-aligned chunks), total ~52 MB.
  char* ws = (char*)d_ws;
  size_t off = 0;
  auto carve = [&](size_t bytes) -> void* {
    void* p = ws + off;
    off += (bytes + 255) & ~(size_t)255;
    return p;
  };
  unsigned short* Wt_hi = (unsigned short*)carve((size_t)H * H * 2);
  unsigned short* Wt_lo = (unsigned short*)carve((size_t)H * H * 2);
  unsigned short* T_hi  = (unsigned short*)carve((size_t)NROWS * H * 2);
  unsigned short* T_lo  = (unsigned short*)carve((size_t)NROWS * H * 2);
  unsigned short* A_hi  = (unsigned short*)carve((size_t)NROWS * H * 2);
  unsigned short* A_lo  = (unsigned short*)carve((size_t)NROWS * H * 2);
  unsigned short* TW_hi = (unsigned short*)carve((size_t)NROWS * H * 2);
  unsigned short* TW_lo = (unsigned short*)carve((size_t)NROWS * H * 2);
  float* tvec = (float*)carve((size_t)NROWS * 4);
  float* avec = (float*)carve((size_t)NROWS * 4);

  // 1) W -> transposed bf16 hi/lo
  convert_w_transpose<<<dim3(H / 32, H / 32), dim3(32, 8), 0, stream>>>(W, Wt_hi, Wt_lo);
  // 2) T/A -> bf16 hi/lo + fused rank-1 dots
  convert_rows_dot<<<NROWS, 256, 0, stream>>>(T, U, T_hi, T_lo, tvec);
  convert_rows_dot<<<NROWS, 256, 0, stream>>>(A, U + H, A_hi, A_lo, avec);
  // 3) TW = T @ W   (B operand = Wt rows), re-split to bf16 hi/lo
  gemm_bf16x3<0><<<dim3(NROWS / 64, H / 256), 256, 0, stream>>>(
      (const __bf16*)T_hi, (const __bf16*)T_lo, H,
      (const __bf16*)Wt_hi, (const __bf16*)Wt_lo, H,
      H, nullptr, H, TW_hi, TW_lo, nullptr, nullptr, nullptr);
  // 4) OUT = TW @ A^T + tvec[i] + avec[j] + b
  gemm_bf16x3<1><<<dim3(NROWS / 64, NROWS / 256), 256, 0, stream>>>(
      (const __bf16*)TW_hi, (const __bf16*)TW_lo, H,
      (const __bf16*)A_hi, (const __bf16*)A_lo, H,
      H, (float*)d_out, NROWS, nullptr, nullptr, tvec, avec, b);
}
